// Attention_7945689497731
// MI455X (gfx1250) — matrix operand order repaired and checked by reading, not compiled, not run on hardware
//
#include <hip/hip_runtime.h>
#include <hip/hip_bf16.h>

typedef __attribute__((ext_vector_type(16))) __bf16 v16bf;
typedef __attribute__((ext_vector_type(8)))  float  v8f;

#define B_      32
#define H_      32
#define D_      128
#define KVH_    8
#define GROUP_  4      // H_/KVH_
#define MAXBLK_ 256
#define BS_     16
#define WAVES_  8
#define NEG_BIG (-1.0e30f)

// pack two float4 (8 consecutive values) into elements [BASE..BASE+7] of a v16bf
#define PUT8(VEC, BASE, A, Bv)                                        \
  do {                                                                \
    (VEC)[(BASE)+0] = (__bf16)(A).x;  (VEC)[(BASE)+1] = (__bf16)(A).y; \
    (VEC)[(BASE)+2] = (__bf16)(A).z;  (VEC)[(BASE)+3] = (__bf16)(A).w; \
    (VEC)[(BASE)+4] = (__bf16)(Bv).x; (VEC)[(BASE)+5] = (__bf16)(Bv).y;\
    (VEC)[(BASE)+6] = (__bf16)(Bv).z; (VEC)[(BASE)+7] = (__bf16)(Bv).w;\
  } while (0)

__global__ __launch_bounds__(256, 1)
void pa_decode_wmma(const float* __restrict__ q,
                    const float* __restrict__ k_cache,
                    const float* __restrict__ v_cache,
                    const int*   __restrict__ block_tables,
                    const int*   __restrict__ ctx_lens,
                    float* __restrict__ out)
{
  const int bkv    = blockIdx.x;          // b * KVH_ + kv
  const int b      = bkv >> 3;
  const int kv     = bkv & 7;
  const int tid    = threadIdx.x;
  const int lane   = tid & 31;
  const int wave   = tid >> 5;
  const int lane15 = lane & 15;
  const int hi     = lane >> 4;           // 0: lanes 0-15, 1: lanes 16-31

  const int ctx     = ctx_lens[b];
  const int nchunks = (ctx + 31) >> 5;    // 32-token chunks (2 cache blocks)
  const int btrow   = b * MAXBLK_;

  __shared__ float sm_m[WAVES_][GROUP_];
  __shared__ float sm_l[WAVES_][GROUP_];
  __shared__ float sm_o[WAVES_][GROUP_][D_];

  // ---- Q in WMMA B-matrix layout (table order: k = 8*half + e, e<8; 16 + 8*half + e-8, e>=8),
  //      pre-scaled by 1/sqrt(D). 4 chunks of K=32 cover D=128. -----------
  const float scale = 0.08838834764831845f;   // 1/sqrt(128)
  v16bf qb[4];
  {
    const int h = lane15;                 // head column (0..3 valid)
    #pragma unroll
    for (int c = 0; c < 4; ++c) {
      if (h < GROUP_) {
        const float* qp = q + (size_t)(b*H_ + kv*GROUP_ + h)*D_ + c*32 + hi*8;
        float4 t0 = *(const float4*)(qp + 0);
        float4 t1 = *(const float4*)(qp + 4);
        float4 t2 = *(const float4*)(qp + 16);
        float4 t3 = *(const float4*)(qp + 20);
        t0.x*=scale; t0.y*=scale; t0.z*=scale; t0.w*=scale;
        t1.x*=scale; t1.y*=scale; t1.z*=scale; t1.w*=scale;
        t2.x*=scale; t2.y*=scale; t2.z*=scale; t2.w*=scale;
        t3.x*=scale; t3.y*=scale; t3.z*=scale; t3.w*=scale;
        PUT8(qb[c], 0, t0, t1);
        PUT8(qb[c], 8, t2, t3);
      } else {
        #pragma unroll
        for (int e = 0; e < 16; ++e) qb[c][e] = (__bf16)0.0f;
      }
    }
  }

  // ---- per-wave online-softmax state --------------------------------------
  float m_run = NEG_BIG, l_run = 0.f;
  v8f Oacc[8];
  #pragma unroll
  for (int cc = 0; cc < 8; ++cc) {
    #pragma unroll
    for (int r = 0; r < 8; ++r) Oacc[cc][r] = 0.f;
  }

  for (int ch = wave; ch < nchunks; ch += WAVES_) {
    const int t0base = ch << 5;
    int blk0 = block_tables[btrow + 2*ch];     blk0 = blk0 < 0 ? 0 : blk0;
    int blk1 = block_tables[btrow + 2*ch + 1]; blk1 = blk1 < 0 ? 0 : blk1;

    // K A-tile row pointers (lane = token row, +hi*8 = documented A k-split)
    const size_t k0row = ((size_t)(blk0*BS_ + lane15)*KVH_ + kv)*(size_t)D_ + hi*8;
    const size_t k1row = ((size_t)(blk1*BS_ + lane15)*KVH_ + kv)*(size_t)D_ + hi*8;
    // V B-tile bases: e<8 -> blk0 row hi*8+e, e>=8 -> blk1 row hi*8+e-8; lane15 = d column
    const size_t vbase0 = ((size_t)(blk0*BS_ + hi*8)*KVH_ + kv)*(size_t)D_ + lane15;
    const size_t vbase1 = ((size_t)(blk1*BS_ + hi*8)*KVH_ + kv)*(size_t)D_ + lane15;

    // prefetch next chunk's K/V rows (global_prefetch_b8)
    {
      const int chn = ch + WAVES_;
      if (chn < nchunks) {
        int nb = block_tables[btrow + 2*chn]; nb = nb < 0 ? 0 : nb;
        const size_t pr = ((size_t)(nb*BS_ + lane15)*KVH_ + kv)*(size_t)D_;
        __builtin_prefetch(k_cache + pr, 0, 3);
        __builtin_prefetch(v_cache + pr, 0, 3);
      }
    }

    // ---- S = K_tile x Q : two 16(tok) x 16(head) f32 tiles ---------------
    v8f s0, s1;
    #pragma unroll
    for (int r = 0; r < 8; ++r) { s0[r] = 0.f; s1[r] = 0.f; }

    #pragma unroll
    for (int c = 0; c < 4; ++c) {
      const float* kp0 = k_cache + k0row + c*32;
      float4 a0 = *(const float4*)(kp0 + 0);
      float4 a1 = *(const float4*)(kp0 + 4);
      float4 a2 = *(const float4*)(kp0 + 16);
      float4 a3 = *(const float4*)(kp0 + 20);
      v16bf aK;
      PUT8(aK, 0, a0, a1);
      PUT8(aK, 8, a2, a3);
      s0 = __builtin_amdgcn_wmma_f32_16x16x32_bf16(false, aK, false, qb[c],
                                                   (short)0, s0, false, false);

      const float* kp1 = k_cache + k1row + c*32;
      float4 b0 = *(const float4*)(kp1 + 0);
      float4 b1 = *(const float4*)(kp1 + 4);
      float4 b2 = *(const float4*)(kp1 + 16);
      float4 b3 = *(const float4*)(kp1 + 20);
      v16bf aK1;
      PUT8(aK1, 0, b0, b1);
      PUT8(aK1, 8, b2, b3);
      s1 = __builtin_amdgcn_wmma_f32_16x16x32_bf16(false, aK1, false, qb[c],
                                                   (short)0, s1, false, false);
    }

    // ---- mask + online softmax (token axis = VGPR row + lane-half) -------
    const int rtok = t0base + hi*8;
    float mx = NEG_BIG;
    #pragma unroll
    for (int r = 0; r < 8; ++r) {
      if (rtok + r      >= ctx) s0[r] = NEG_BIG;
      if (rtok + 16 + r >= ctx) s1[r] = NEG_BIG;
      mx = fmaxf(mx, fmaxf(s0[r], s1[r]));
    }
    mx = fmaxf(mx, __shfl_xor(mx, 16, 32));       // lanes L <-> L+16: same head
    const float m_new = fmaxf(m_run, mx);
    const float alpha = __expf(m_run - m_new);    // first iter: underflows to 0
    float sum = 0.f;
    #pragma unroll
    for (int r = 0; r < 8; ++r) {
      s0[r] = __expf(s0[r] - m_new);
      s1[r] = __expf(s1[r] - m_new);
      sum += s0[r] + s1[r];
    }
    sum += __shfl_xor(sum, 16, 32);
    l_run = l_run * alpha + sum;
    m_run = m_new;

    // rescale O: row r of D-tile is head r (lanes<16); fetch alpha from lane r
    #pragma unroll
    for (int r = 0; r < 8; ++r) {
      const float ar = __shfl(alpha, r, 32);
      #pragma unroll
      for (int cc = 0; cc < 8; ++cc) Oacc[cc][r] *= ar;
    }

    // C-layout of S^T == A-layout of P : pure in-lane bf16 repack
    v16bf pA;
    #pragma unroll
    for (int e = 0; e < 8; ++e) {
      pA[e]     = (__bf16)s0[e];
      pA[e + 8] = (__bf16)s1[e];
    }

    // ---- O += P x V : 8 d-chunks of 16 -----------------------------------
    #pragma unroll
    for (int cc = 0; cc < 8; ++cc) {
      v16bf vb;
      #pragma unroll
      for (int e = 0; e < 16; ++e)
        vb[e] = (__bf16)v_cache[(e < 8 ? vbase0 + (size_t)e*(KVH_*D_) : vbase1 + (size_t)(e-8)*(KVH_*D_)) + cc*16];
      Oacc[cc] = __builtin_amdgcn_wmma_f32_16x16x32_bf16(false, pA, false, vb,
                                                         (short)0, Oacc[cc],
                                                         false, false);
    }
  }

  // ---- publish per-wave partials (heads 0..3 live in VGPR rows 0..3,
  //      lanes 0..15; d = chunk*16 + lane) ---------------------------------
  if (lane < 16) {
    #pragma unroll
    for (int cc = 0; cc < 8; ++cc) {
      #pragma unroll
      for (int r = 0; r < GROUP_; ++r)
        sm_o[wave][r][cc*16 + lane] = Oacc[cc][r];
    }
  }
  if (lane < GROUP_) {                 // lane == head for lanes 0..3
    sm_m[wave][lane] = m_run;
    sm_l[wave][lane] = l_run;
  }
  __syncthreads();

  // ---- cross-wave flash merge + normalize; 512 outputs, 256 threads -----
  for (int e = tid; e < GROUP_ * D_; e += blockDim.x) {
    const int hh = e >> 7;
    const int d  = e & (D_ - 1);
    float M = NEG_BIG;
    #pragma unroll
    for (int w = 0; w < WAVES_; ++w) M = fmaxf(M, sm_m[w][hh]);
    float L = 0.f, acc = 0.f;
    #pragma unroll
    for (int w = 0; w < WAVES_; ++w) {
      const float sc = __expf(sm_m[w][hh] - M);   // empty wave -> 0
      L   += sm_l[w][hh] * sc;
      acc += sm_o[w][hh][d] * sc;
    }
    out[(size_t)(b*H_ + kv*GROUP_ + hh)*D_ + d] = acc / L;
  }
}

extern "C" void kernel_launch(void* const* d_in, const int* in_sizes, int n_in,
                              void* d_out, int out_size, void* d_ws, size_t ws_size,
                              hipStream_t stream) {
  (void)in_sizes; (void)n_in; (void)out_size; (void)d_ws; (void)ws_size;
  const float* q            = (const float*)d_in[0];
  const float* k_cache      = (const float*)d_in[1];
  const float* v_cache      = (const float*)d_in[2];
  const int*   block_tables = (const int*)d_in[3];
  const int*   context_lens = (const int*)d_in[4];
  float* out = (float*)d_out;

  dim3 grid(B_ * KVH_);        // 256 workgroups: one per (batch, kv head)
  dim3 block(WAVES_ * 32);     // 8 wave32s
  pa_decode_wmma<<<grid, block, 0, stream>>>(q, k_cache, v_cache,
                                             block_tables, context_lens, out);
}
